// PVCNNVelocityNet_9053791060443
// MI455X (gfx1250) — matrix operand order repaired and checked by reading, not compiled, not run on hardware
//
#include <hip/hip_runtime.h>
#include <hip/hip_bf16.h>

// ---------------- problem constants ----------------
#define BB 2
#define NP 8192
#define EMB 256
#define CH 128
#define RR 32
#define LL 3
#define RCUBE 32768           // 32^3
#define BN1_EPS 1e-5f
#define BN3_EPS 1e-4f
#define VOX_EPS 1e-6f

// conv input tile in LDS: [col(9)][z(18)][ci pitch 144]
#define CPITCH 144
#define CONV_TILE_ELEMS (9 * 18 * CH)       // 20736 staged values
#define CONV_LDS_HALVES (9 * 18 * CPITCH)   // 23328 halves = 46656 B

// packed weight sizes (halves)
#define CONVW_PACK (27 * 4 * 8 * 32 * 16)   // 442368 per conv layer
#define MLPW_PACK  (4 * 8 * 32 * 16)        // 16384 per 128x128 matrix

typedef __attribute__((ext_vector_type(16))) _Float16 v16h;
typedef __attribute__((ext_vector_type(8)))  float    v8f;

// ---------------------------------------------------------------------------
// Weight pre-pack kernels: rearrange f32 weights into this code's contiguous-k WMMA
// 16-bit A-matrix (16x32) per-lane fragment layout (matching the B loads' k order), so the GEMM inner loop
// loads each fragment with two coalesced b128 loads.
// A(m,k): lane = m + 16*lh where lanes hold K in {lh*16..lh*16+15};
// fragment half i -> k = lh*16+i.
// ---------------------------------------------------------------------------
__device__ __forceinline__ int frag_k(int lh, int i) {
    return lh * 16 + i;
}

__global__ __launch_bounds__(256) void pack_conv_w_kernel(
    const float* __restrict__ W,   // (128,128,27)
    _Float16* __restrict__ Wp)     // [o][cc][tile8][lane32][i16]
{
    int idx = blockIdx.x * 256 + threadIdx.x;
    if (idx >= CONVW_PACK) return;
    int i    = idx & 15;
    int lane = (idx >> 4) & 31;
    int tile = (idx >> 9) & 7;
    int cc   = (idx >> 12) & 3;
    int o    = idx >> 14;
    int m = lane & 15, lh = lane >> 4;
    int ci = cc * 32 + frag_k(lh, i);
    int co = tile * 16 + m;
    Wp[idx] = (_Float16)W[((size_t)(co * CH + ci)) * 27 + o];
}

__global__ __launch_bounds__(256) void pack_mlp_w_kernel(
    const float* __restrict__ W,   // (128,128)
    _Float16* __restrict__ Wp)     // [cc][tile8][lane32][i16]
{
    int idx = blockIdx.x * 256 + threadIdx.x;
    if (idx >= MLPW_PACK) return;
    int i    = idx & 15;
    int lane = (idx >> 4) & 31;
    int tile = (idx >> 9) & 7;
    int cc   = idx >> 12;
    int m = lane & 15, lh = lane >> 4;
    int ci = cc * 32 + frag_k(lh, i);
    int co = tile * 16 + m;
    Wp[idx] = (_Float16)W[(size_t)co * CH + ci];
}

// ---------------------------------------------------------------------------
// emb kernel: sinusoidal time embedding + projections + silu; fold the 256
// constant embedding channels of the stem into base[b][c].
// ---------------------------------------------------------------------------
__global__ __launch_bounds__(256) void emb_kernel(
    const float* __restrict__ t, const float* __restrict__ cond,
    const float* __restrict__ tw, const float* __restrict__ tb,
    const float* __restrict__ cw, const float* __restrict__ cb,
    const float* __restrict__ stem_w, const float* __restrict__ stem_b,
    float* __restrict__ emb, float* __restrict__ base)
{
    int b = blockIdx.x;
    int tid = threadIdx.x;
    __shared__ float temb[EMB];
    __shared__ float e[EMB];
    if (tid < EMB / 2) {
        float fr  = __expf(-logf(10000.0f) * (float)tid / (float)(EMB / 2));
        float arg = t[b] * fr;
        temb[tid]           = cosf(arg);
        temb[tid + EMB / 2] = sinf(arg);
    }
    __syncthreads();
    float s1 = tb[tid];
    for (int i = 0; i < EMB; ++i) s1 += temb[i] * tw[i * EMB + tid];
    float s2 = cb[tid];
    for (int i = 0; i < CH; ++i)  s2 += cond[b * CH + i] * cw[i * EMB + tid];
    float v = s1 / (1.0f + expf(-s1)) + s2 / (1.0f + expf(-s2));
    e[tid] = v;
    emb[b * EMB + tid] = v;
    __syncthreads();
    if (tid < CH) {
        float s = stem_b[tid];
        const float* wr = stem_w + (size_t)tid * (EMB + 3);
        for (int i = 0; i < EMB; ++i) s += wr[i] * e[i];
        base[b * CH + tid] = s;
    }
}

// ---------------------------------------------------------------------------
// voxel-coord kernel: mean-center, scale by 2*max-norm, map to [0, R-1].
// ---------------------------------------------------------------------------
__global__ __launch_bounds__(256) void voxcoord_kernel(
    const float* __restrict__ x, float* __restrict__ nc)
{
    int b = blockIdx.x;
    int tid = threadIdx.x;
    __shared__ float red[3 * 256];
    __shared__ float meanS[3];
    __shared__ float mnorm;

    float sx = 0.f, sy = 0.f, sz = 0.f;
    for (int n = tid; n < NP; n += 256) {
        const float* p = x + (size_t)(b * NP + n) * 3;
        sx += p[0]; sy += p[1]; sz += p[2];
    }
    red[tid] = sx; red[256 + tid] = sy; red[512 + tid] = sz;
    __syncthreads();
    for (int st = 128; st > 0; st >>= 1) {
        if (tid < st) {
            red[tid]       += red[tid + st];
            red[256 + tid] += red[256 + tid + st];
            red[512 + tid] += red[512 + tid + st];
        }
        __syncthreads();
    }
    if (tid == 0) {
        meanS[0] = red[0]   / (float)NP;
        meanS[1] = red[256] / (float)NP;
        meanS[2] = red[512] / (float)NP;
    }
    __syncthreads();
    float m0 = meanS[0], m1 = meanS[1], m2 = meanS[2];
    float mx = 0.f;
    for (int n = tid; n < NP; n += 256) {
        const float* p = x + (size_t)(b * NP + n) * 3;
        float cx = p[0] - m0, cy = p[1] - m1, cz = p[2] - m2;
        mx = fmaxf(mx, sqrtf(cx * cx + cy * cy + cz * cz));
    }
    red[tid] = mx;
    __syncthreads();
    for (int st = 128; st > 0; st >>= 1) {
        if (tid < st) red[tid] = fmaxf(red[tid], red[tid + st]);
        __syncthreads();
    }
    if (tid == 0) mnorm = red[0];
    __syncthreads();
    float inv = 1.0f / (mnorm * 2.0f + VOX_EPS);
    for (int n = tid; n < NP; n += 256) {
        const float* p = x + (size_t)(b * NP + n) * 3;
        float c[3] = { p[0] - m0, p[1] - m1, p[2] - m2 };
        #pragma unroll
        for (int j = 0; j < 3; ++j) {
            float v = (c[j] * inv + 0.5f) * (float)RR;
            nc[((size_t)(b * 3 + j)) * NP + n] = fminf(fmaxf(v, 0.0f), (float)(RR - 1));
        }
    }
}

// ---------------------------------------------------------------------------
// stem: f[b][c][n] = relu(bn1(base[b][c] + W_coord . coords))
// ---------------------------------------------------------------------------
__global__ __launch_bounds__(256) void stem_kernel(
    const float* __restrict__ x, const float* __restrict__ base,
    const float* __restrict__ stem_w, const float* __restrict__ g,
    const float* __restrict__ beta, float* __restrict__ f)
{
    int idx = blockIdx.x * 256 + threadIdx.x;   // B*CH*NP = 2^21
    if (idx >= BB * CH * NP) return;
    int n = idx & (NP - 1);
    int c = (idx >> 13) & (CH - 1);
    int b = idx >> 20;
    const float* xr = x + (size_t)(b * NP + n) * 3;
    const float* wr = stem_w + (size_t)c * (EMB + 3) + EMB;
    float v = base[b * CH + c] + wr[0] * xr[0] + wr[1] * xr[1] + wr[2] * xr[2];
    v = v * (g[c] * rsqrtf(1.0f + BN1_EPS)) + beta[c];
    f[idx] = fmaxf(v, 0.0f);
}

// ---------------------------------------------------------------------------
// scatter-add points into voxel grid (+ counts)
// ---------------------------------------------------------------------------
__global__ __launch_bounds__(256) void scatter_kernel(
    const float* __restrict__ f, const float* __restrict__ nc,
    float* __restrict__ gsum, float* __restrict__ cnt)
{
    int idx = blockIdx.x * 256 + threadIdx.x;   // B*NP
    if (idx >= BB * NP) return;
    int b = idx >> 13, n = idx & (NP - 1);
    int vx = (int)rintf(nc[((size_t)(b * 3 + 0)) * NP + n]);
    int vy = (int)rintf(nc[((size_t)(b * 3 + 1)) * NP + n]);
    int vz = (int)rintf(nc[((size_t)(b * 3 + 2)) * NP + n]);
    int v  = (vx * RR + vy) * RR + vz;
    atomicAdd(&cnt[((size_t)b << 15) + v], 1.0f);
    for (int c = 0; c < CH; ++c)
        atomicAdd(&gsum[(((size_t)(b * CH + c)) << 15) + v],
                  f[((size_t)(b * CH + c)) * NP + n]);
}

__global__ __launch_bounds__(256) void vox_norm_kernel(
    float* __restrict__ gsum, const float* __restrict__ cnt)
{
    int idx = blockIdx.x * 256 + threadIdx.x;   // B*CH*RCUBE = 2^23
    if (idx >= BB * CH * RCUBE) return;
    int v = idx & (RCUBE - 1);
    int b = idx >> 22;
    float ct = cnt[((size_t)b << 15) + v];
    gsum[idx] *= 1.0f / fmaxf(ct, 1.0f);
}

// ---------------------------------------------------------------------------
// 3x3x3 Conv3d (SAME) + bias + BN3d + LeakyReLU(0.1), implicit GEMM on WMMA.
// Block = 128 threads (4 waves). Wave w computes out-channels [32w, 32w+32)
// for 16 z-consecutive voxels (two 16x16 WMMA tiles sharing one B fragment).
// Entire 3x3 xy-halo x 18z x 128ci input tile staged to LDS once (f16,
// [col][z][ci] with pitch 144 for bank spread); the 108-chunk K loop then
// runs purely out of LDS (b128 fragment loads) + pre-packed b128 A loads.
// ---------------------------------------------------------------------------
__global__ __launch_bounds__(128) void conv3d_wmma_kernel(
    const float* __restrict__ gin, float* __restrict__ gout,
    const _Float16* __restrict__ Wp,   // packed [o][cc][tile8][lane32][i16]
    const float* __restrict__ bias, const float* __restrict__ bng,
    const float* __restrict__ bnb)
{
    __shared__ _Float16 Bt[CONV_LDS_HALVES];
    int tid  = threadIdx.x;
    int lane = tid & 31;
    int wave = tid >> 5;
    int blk  = blockIdx.x;            // B * (RCUBE/16)
    int b    = blk >> 11;
    int tile = blk & 2047;
    int vox0 = tile << 4;
    int gx = vox0 >> 10;
    int gy = (vox0 >> 5) & 31;
    int z0 = vox0 & 31;               // 0 or 16

    // ---- stage full input tile: ci x 9 cols x 18 z (z innermost: coalesced)
    const float* ginb = gin + (((size_t)b * CH) << 15);
    for (int e = tid; e < CONV_TILE_ELEMS; e += 128) {
        int ci  = e / 162;            // 162 = 9*18
        int rem = e - ci * 162;
        int col = rem / 18;
        int z   = rem - col * 18;
        int xi = gx + col / 3 - 1;
        int yi = gy + col % 3 - 1;
        int zi = z0 + z - 1;
        float v = 0.0f;
        if (((unsigned)xi < (unsigned)RR) & ((unsigned)yi < (unsigned)RR) &
            ((unsigned)zi < (unsigned)RR))
            v = ginb[((size_t)ci << 15) + (size_t)((xi * RR + yi) * RR + zi)];
        Bt[(col * 18 + z) * CPITCH + ci] = (_Float16)v;
    }
    __syncthreads();

    int n  = lane & 15;               // voxel within tile / A row m
    int lh = lane >> 4;

    v8f acc0 = {};
    v8f acc1 = {};

    for (int o = 0; o < 27; ++o) {
        int col  = o / 3;             // (dx+1)*3 + (dy+1)
        int zidx = n + (o % 3);       // local z index (0..17)
        const _Float16* wrow = Wp + (size_t)o * (4 * 8 * 32 * 16);
        if (o + 1 < 27) __builtin_prefetch(wrow + 4 * 8 * 32 * 16, 0, 1);
        const _Float16* brow = Bt + (col * 18 + zidx) * CPITCH + lh * 16;
        #pragma unroll
        for (int cc = 0; cc < 4; ++cc) {
            v16h af0 = *(const v16h*)(wrow + (((cc * 8) + 2 * wave)     * 32 + lane) * 16);
            v16h af1 = *(const v16h*)(wrow + (((cc * 8) + 2 * wave + 1) * 32 + lane) * 16);
            v16h bf  = *(const v16h*)(brow + cc * 32);
            acc0 = __builtin_amdgcn_wmma_f32_16x16x32_f16(
                false, af0, false, bf, (short)0, acc0, false, false);
            acc1 = __builtin_amdgcn_wmma_f32_16x16x32_f16(
                false, af1, false, bf, (short)0, acc1, false, false);
        }
    }

    // ---- epilogue: bias + BN3d + leaky relu
    float bscale = rsqrtf(1.0f + BN3_EPS);
    int coBase = wave << 5;
    #pragma unroll
    for (int v = 0; v < 8; ++v) {
        int m   = v + (lh << 3);
        int co0 = coBase + m;
        int co1 = co0 + 16;
        float y0 = acc0[v] + bias[co0];
        y0 = y0 * (bng[co0] * bscale) + bnb[co0];
        y0 = (y0 > 0.0f) ? y0 : 0.1f * y0;
        gout[(((size_t)(b * CH + co0)) << 15) + (size_t)(vox0 + n)] = y0;
        float y1 = acc1[v] + bias[co1];
        y1 = y1 * (bng[co1] * bscale) + bnb[co1];
        y1 = (y1 > 0.0f) ? y1 : 0.1f * y1;
        gout[(((size_t)(b * CH + co1)) << 15) + (size_t)(vox0 + n)] = y1;
    }
}

// ---------------------------------------------------------------------------
// Pointwise MLP: Y = relu(bn1(W X + bias)) (+ addIn), W 128x128, on WMMA.
// Block = 256 threads (8 waves); block covers 128 out-ch x 16 points.
// X tile (128x16) staged once to LDS as [n][ci] (pitch 144).
// ---------------------------------------------------------------------------
__global__ __launch_bounds__(256) void pointmlp_wmma_kernel(
    const float* __restrict__ X, const _Float16* __restrict__ Wp,
    const float* __restrict__ bias, const float* __restrict__ bng,
    const float* __restrict__ bnb, const float* __restrict__ addIn,
    float* __restrict__ Y)
{
    __shared__ _Float16 Xs[16 * CPITCH];
    int tid  = threadIdx.x;
    int lane = tid & 31;
    int wave = tid >> 5;
    int blk  = blockIdx.x;            // B * (NP/16)
    int b    = blk >> 9;
    int n0   = (blk & 511) << 4;

    for (int e = tid; e < CH * 16; e += 256) {
        int ci = e >> 4;
        int nn = e & 15;
        Xs[nn * CPITCH + ci] =
            (_Float16)X[((size_t)(b * CH + ci)) * NP + n0 + nn];
    }
    __syncthreads();

    int n  = lane & 15;
    int lh = lane >> 4;

    v8f acc = {};
    const _Float16* brow = Xs + n * CPITCH + lh * 16;
    #pragma unroll
    for (int cc = 0; cc < 4; ++cc) {
        v16h af = *(const v16h*)(Wp + (((cc * 8) + wave) * 32 + lane) * 16);
        v16h bf = *(const v16h*)(brow + cc * 32);
        acc = __builtin_amdgcn_wmma_f32_16x16x32_f16(
            false, af, false, bf, (short)0, acc, false, false);
    }

    float bscale = rsqrtf(1.0f + BN1_EPS);
    int coBase = wave << 4;
    #pragma unroll
    for (int v = 0; v < 8; ++v) {
        int m  = v + (lh << 3);
        int co = coBase + m;
        size_t oidx = ((size_t)(b * CH + co)) * NP + n0 + n;
        float y = acc[v] + bias[co];
        y = y * (bng[co] * bscale) + bnb[co];
        y = fmaxf(y, 0.0f);
        if (addIn) y += addIn[oidx];
        Y[oidx] = y;
    }
}

// ---------------------------------------------------------------------------
// SE squeeze: mean over voxels per (b,c)
// ---------------------------------------------------------------------------
__global__ __launch_bounds__(256) void se_reduce_kernel(
    const float* __restrict__ g2, float* __restrict__ s)
{
    int bc = blockIdx.x;              // B*CH
    int tid = threadIdx.x;
    __shared__ float red[256];
    float acc = 0.0f;
    const float* gp = g2 + ((size_t)bc << 15);
    for (int v = tid; v < RCUBE; v += 256) acc += gp[v];
    red[tid] = acc;
    __syncthreads();
    for (int st = 128; st > 0; st >>= 1) {
        if (tid < st) red[tid] += red[tid + st];
        __syncthreads();
    }
    if (tid == 0) s[bc] = red[0] / (float)RCUBE;
}

__global__ __launch_bounds__(128) void se_mlp_kernel(
    const float* __restrict__ s, const float* __restrict__ w1,
    const float* __restrict__ b1, const float* __restrict__ w2,
    const float* __restrict__ b2, float* __restrict__ a)
{
    int b = blockIdx.x;
    int tid = threadIdx.x;            // 128
    __shared__ float h[CH / 8];
    if (tid < CH / 8) {
        float v = b1[tid];
        for (int c = 0; c < CH; ++c) v += s[b * CH + c] * w1[c * (CH / 8) + tid];
        h[tid] = fmaxf(v, 0.0f);
    }
    __syncthreads();
    float v = b2[tid];
    for (int j = 0; j < CH / 8; ++j) v += h[j] * w2[j * CH + tid];
    a[b * CH + tid] = 1.0f / (1.0f + expf(-v));
}

// ---------------------------------------------------------------------------
// trilinear devoxelize with SE scale folded in
// ---------------------------------------------------------------------------
__global__ __launch_bounds__(128) void devox_kernel(
    const float* __restrict__ g2, const float* __restrict__ a,
    const float* __restrict__ nc, float* __restrict__ out)
{
    int bn = blockIdx.x;              // B*NP
    int b = bn >> 13, n = bn & (NP - 1);
    int c = threadIdx.x;
    float px = nc[((size_t)(b * 3 + 0)) * NP + n];
    float py = nc[((size_t)(b * 3 + 1)) * NP + n];
    float pz = nc[((size_t)(b * 3 + 2)) * NP + n];
    float fx = floorf(px), fy = floorf(py), fz = floorf(pz);
    int lx = (int)fx, ly = (int)fy, lz = (int)fz;
    int hx = min(lx + 1, RR - 1), hy = min(ly + 1, RR - 1), hz = min(lz + 1, RR - 1);
    float tx = px - fx, ty = py - fy, tz = pz - fz;
    const float* gc = g2 + (((size_t)(b * CH + c)) << 15);
    float acc = 0.0f;
    #pragma unroll
    for (int dx = 0; dx < 2; ++dx) {
        int ix = dx ? hx : lx;
        float wx = dx ? tx : 1.0f - tx;
        #pragma unroll
        for (int dy = 0; dy < 2; ++dy) {
            int iy = dy ? hy : ly;
            float wy = dy ? ty : 1.0f - ty;
            #pragma unroll
            for (int dz = 0; dz < 2; ++dz) {
                int iz = dz ? hz : lz;
                float wz = dz ? tz : 1.0f - tz;
                acc += wx * wy * wz * gc[(ix * RR + iy) * RR + iz];
            }
        }
    }
    out[((size_t)(b * CH + c)) * NP + n] = acc * a[b * CH + c];
}

// ---------------------------------------------------------------------------
// final projection: out[b][n][o] = out_w[o] . feat[b][:][n] + out_b[o]
// ---------------------------------------------------------------------------
__global__ __launch_bounds__(256) void out_kernel(
    const float* __restrict__ feat, const float* __restrict__ w,
    const float* __restrict__ bo, float* __restrict__ out)
{
    int idx = blockIdx.x * 256 + threadIdx.x;   // B*NP*3
    if (idx >= BB * NP * 3) return;
    int o = idx % 3;
    int n = (idx / 3) & (NP - 1);
    int b = idx / (3 * NP);
    float v = bo[o];
    const float* wr = w + o * CH;
    for (int c = 0; c < CH; ++c)
        v += wr[c] * feat[((size_t)(b * CH + c)) * NP + n];
    out[idx] = v;
}

// ---------------------------------------------------------------------------
extern "C" void kernel_launch(void* const* d_in, const int* in_sizes, int n_in,
                              void* d_out, int out_size, void* d_ws, size_t ws_size,
                              hipStream_t stream) {
    const float* x        = (const float*)d_in[0];
    const float* t        = (const float*)d_in[1];
    const float* cond     = (const float*)d_in[2];
    const float* t_proj_w = (const float*)d_in[3];
    const float* t_proj_b = (const float*)d_in[4];
    const float* c_proj_w = (const float*)d_in[5];
    const float* c_proj_b = (const float*)d_in[6];
    const float* stem_w   = (const float*)d_in[7];
    const float* stem_b   = (const float*)d_in[8];
    const float* stem_g   = (const float*)d_in[9];
    const float* stem_be  = (const float*)d_in[10];
    const float* c1w = (const float*)d_in[11];
    const float* c1b = (const float*)d_in[12];
    const float* g1g = (const float*)d_in[13];
    const float* g1b = (const float*)d_in[14];
    const float* c2w = (const float*)d_in[15];
    const float* c2b = (const float*)d_in[16];
    const float* g2g = (const float*)d_in[17];
    const float* g2b = (const float*)d_in[18];
    const float* se1w = (const float*)d_in[19];
    const float* se1b = (const float*)d_in[20];
    const float* se2w = (const float*)d_in[21];
    const float* se2b = (const float*)d_in[22];
    const float* pfw = (const float*)d_in[23];
    const float* pfb = (const float*)d_in[24];
    const float* pfg = (const float*)d_in[25];
    const float* pfbe = (const float*)d_in[26];
    const float* mw = (const float*)d_in[27];
    const float* mb = (const float*)d_in[28];
    const float* mg = (const float*)d_in[29];
    const float* mbe = (const float*)d_in[30];
    const float* hw = (const float*)d_in[31];
    const float* hb = (const float*)d_in[32];
    const float* hg = (const float*)d_in[33];
    const float* hbe = (const float*)d_in[34];
    const float* ow = (const float*)d_in[35];
    const float* ob = (const float*)d_in[36];
    float* out = (float*)d_out;

    // ---- workspace carve-up (floats, then halves) ----
    float* ws = (float*)d_ws;
    size_t off = 0;
    float* emb  = ws + off; off += (size_t)BB * EMB;
    float* base = ws + off; off += (size_t)BB * CH;
    float* nc   = ws + off; off += (size_t)BB * 3 * NP;
    float* f    = ws + off; off += (size_t)BB * CH * NP;
    float* f2   = ws + off; off += (size_t)BB * CH * NP;
    float* dvx  = ws + off; off += (size_t)BB * CH * NP;
    float* g0   = ws + off; off += (size_t)BB * CH * RCUBE;
    float* g1   = ws + off; off += (size_t)BB * CH * RCUBE;
    float* cnt  = ws + off; off += (size_t)BB * RCUBE;
    float* sbuf = ws + off; off += (size_t)BB * CH;
    float* abuf = ws + off; off += (size_t)BB * CH;
    off = (off + 63) & ~(size_t)63;                      // 32B+ align
    _Float16* hws = (_Float16*)(ws + off);
    size_t hoff = 0;
    _Float16* c1p = hws + hoff; hoff += (size_t)LL * CONVW_PACK;   // conv1 x3
    _Float16* c2p = hws + hoff; hoff += (size_t)LL * CONVW_PACK;   // conv2 x3
    _Float16* pfp = hws + hoff; hoff += (size_t)LL * MLPW_PACK;
    _Float16* mlp = hws + hoff; hoff += (size_t)LL * MLPW_PACK;
    _Float16* hdp = hws + hoff; hoff += (size_t)MLPW_PACK;
    (void)ws_size; (void)in_sizes; (void)n_in; (void)out_size;

    // ---- weight pre-packing into WMMA fragment layout ----
    for (int l = 0; l < LL; ++l) {
        pack_conv_w_kernel<<<(CONVW_PACK + 255) / 256, 256, 0, stream>>>(
            c1w + (size_t)l * CH * CH * 27, c1p + (size_t)l * CONVW_PACK);
        pack_conv_w_kernel<<<(CONVW_PACK + 255) / 256, 256, 0, stream>>>(
            c2w + (size_t)l * CH * CH * 27, c2p + (size_t)l * CONVW_PACK);
        pack_mlp_w_kernel<<<(MLPW_PACK + 255) / 256, 256, 0, stream>>>(
            pfw + (size_t)l * CH * CH, pfp + (size_t)l * MLPW_PACK);
        pack_mlp_w_kernel<<<(MLPW_PACK + 255) / 256, 256, 0, stream>>>(
            mw + (size_t)l * CH * CH, mlp + (size_t)l * MLPW_PACK);
    }
    pack_mlp_w_kernel<<<(MLPW_PACK + 255) / 256, 256, 0, stream>>>(hw, hdp);

    // ---- embeddings / voxel coords / stem ----
    emb_kernel<<<BB, 256, 0, stream>>>(t, cond, t_proj_w, t_proj_b,
                                       c_proj_w, c_proj_b, stem_w, stem_b, emb, base);
    voxcoord_kernel<<<BB, 256, 0, stream>>>(x, nc);
    stem_kernel<<<(BB * CH * NP) / 256, 256, 0, stream>>>(x, base, stem_w,
                                                          stem_g, stem_be, f);

    for (int l = 0; l < LL; ++l) {
        // voxelize (scatter-mean)
        hipMemsetAsync(g0, 0, (size_t)BB * CH * RCUBE * sizeof(float), stream);
        hipMemsetAsync(cnt, 0, (size_t)BB * RCUBE * sizeof(float), stream);
        scatter_kernel<<<(BB * NP + 255) / 256, 256, 0, stream>>>(f, nc, g0, cnt);
        vox_norm_kernel<<<(BB * CH * RCUBE) / 256, 256, 0, stream>>>(g0, cnt);

        // conv1 + bn + leaky   g0 -> g1
        conv3d_wmma_kernel<<<BB * (RCUBE / 16), 128, 0, stream>>>(
            g0, g1, c1p + (size_t)l * CONVW_PACK, c1b + l * CH,
            g1g + l * CH, g1b + l * CH);
        // conv2 + bn + leaky   g1 -> g0  (g0 now holds g2)
        conv3d_wmma_kernel<<<BB * (RCUBE / 16), 128, 0, stream>>>(
            g1, g0, c2p + (size_t)l * CONVW_PACK, c2b + l * CH,
            g2g + l * CH, g2b + l * CH);

        // SE
        se_reduce_kernel<<<BB * CH, 256, 0, stream>>>(g0, sbuf);
        se_mlp_kernel<<<BB, 128, 0, stream>>>(sbuf, se1w + (size_t)l * CH * (CH / 8),
                                              se1b + l * (CH / 8),
                                              se2w + (size_t)l * (CH / 8) * CH,
                                              se2b + l * CH, abuf);

        // devoxelize (SE scale folded in)
        devox_kernel<<<BB * NP, 128, 0, stream>>>(g0, abuf, nc, dvx);

        // point branch
        pointmlp_wmma_kernel<<<BB * (NP / 16), 256, 0, stream>>>(
            f, pfp + (size_t)l * MLPW_PACK, pfb + l * CH,
            pfg + l * CH, pfbe + l * CH, dvx, f2);
        pointmlp_wmma_kernel<<<BB * (NP / 16), 256, 0, stream>>>(
            f2, mlp + (size_t)l * MLPW_PACK, mb + l * CH,
            mg + l * CH, mbe + l * CH, nullptr, f);
    }

    // head + output projection
    pointmlp_wmma_kernel<<<BB * (NP / 16), 256, 0, stream>>>(
        f, hdp, hb, hg, hbe, nullptr, f2);
    out_kernel<<<(BB * NP * 3 + 255) / 256, 256, 0, stream>>>(f2, ow, ob, out);
}
